// DisturbanceRegressionLoss_41446434406829
// MI455X (gfx1250) — matrix operand order repaired and checked by reading, not compiled, not run on hardware
//
#include <hip/hip_runtime.h>

// DisturbanceRegressionLoss for MI455X (gfx1250, wave32).
// out: (B=16, Y=32, H=256, W=256) f32. target unused. Output: 1 f32 (mean loss).
// Memory-bound (~128 MiB read @ 23.3 TB/s ~ 6 us). Single fused streaming pass.
// The reference einsums ('ft,bthw->fbhw') are realized as WMMA matmuls:
//   P0 = M1 x V, P1 = M2 x V, P2 = M1 x (V*V), with exact hi/lo f16 splitting.

typedef __attribute__((ext_vector_type(16))) _Float16 v16h;
typedef __attribute__((ext_vector_type(8)))  float    v8f;

#define YD 32
#define HW 65536                      // H*W = 256*256
#define NPIX (16 * 65536)             // B*H*W = 1,048,576 pixels
#define WAVES_PER_BLOCK 4
#define BLOCK (WAVES_PER_BLOCK * 32)  // 128 threads = 4 wave32
#define GRID 512
#define ITERS (NPIX / (GRID * WAVES_PER_BLOCK * 32))  // 16 tiles of 32 pixels per wave

__global__ void drl_zero_kernel(float* p) {
  if (threadIdx.x == 0) p[0] = 0.0f;
}

__global__ __launch_bounds__(BLOCK)
void drl_main_kernel(const float* __restrict__ outp, float* __restrict__ lossp) {
  const int lane = threadIdx.x & 31;
  const int wib  = threadIdx.x >> 5;                       // wave in block
  const int gw   = (blockIdx.x * BLOCK + threadIdx.x) >> 5; // global wave id
  const int half = lane >> 4;                              // lane half (0/1)
  const int nlo  = lane & 15;

  // Per-wave staging of WMMA outputs: sP[wave][P0/P1/P2][f=0..31][pixel=0..31]
  __shared__ float sP[WAVES_PER_BLOCK][3][32][32];         // 48 KB

  // ---- A fragments (mask matrices), loop invariant ----
  // A is 16x32 f16. Lane layout (matches B): lanes 0-15 row M=nlo hold
  // K = 0..15; lanes 16-31 same row hold K = 16..31.
  // M1[f][K] = (K < f), M2[f][K] = K * (K < f). Exact in f16 (K <= 31).
  v16h a1[2], a2[2];
#pragma unroll
  for (int e = 0; e < 16; ++e) {
    const int kA = 16 * half + e;
#pragma unroll
    for (int ft = 0; ft < 2; ++ft) {
      const int f = nlo + 16 * ft;                 // f row of this tile
      const float m1 = (kA < f) ? 1.0f : 0.0f;
      a1[ft][e] = (_Float16)m1;
      a2[ft][e] = (_Float16)(m1 * (float)kA);
    }
  }

  float acc = 0.0f;

  for (int it = 0; it < ITERS; ++it) {
    // 32 contiguous pixels per wave tile (contiguous in w -> coalesced loads)
    const int q    = (gw * ITERS + it) * 32 + lane;        // pixel id (b*HW + hw)
    const int base = ((q >> 16) << 21) + (q & (HW - 1));   // b*Y*HW + hw

    // ---- Load full Y column into registers (128B/wave per y-slice) ----
    float v[YD];
#pragma unroll
    for (int y = 0; y < YD; ++y) v[y] = outp[base + y * HW];

    // Prefetch next tile (global_prefetch_b8), hide HBM latency
    if (it + 1 < ITERS) {
      const int q2 = q + 32;
      const int b2 = ((q2 >> 16) << 21) + (q2 & (HW - 1));
#pragma unroll
      for (int y = 0; y < YD; y += 4) __builtin_prefetch(&outp[b2 + y * HW], 0, 1);
    }

    // ---- argmin of diff: d[0]=-7, d[1]=d[31]=0, else v[y]-v[y-1]; first min ----
    float best = -7.0f;
    int fidx = 0;
#pragma unroll
    for (int y = 1; y < YD; ++y) {
      const float d = (y == 1 || y == YD - 1) ? 0.0f : (v[y] - v[y - 1]);
      if (d < best) { best = d; fidx = y; }
    }

    // ---- totals S0 = sum v, S1 = sum t*v, S2 = sum v^2 ----
    float S0 = 0.0f, S1 = 0.0f, S2 = 0.0f;
#pragma unroll
    for (int y = 0; y < YD; ++y) {
      S0 += v[y];
      S1 += (float)y * v[y];
      S2 += v[y] * v[y];
    }

    // Other half's pixel column (for B-fragment assembly)
    float u[YD];
#pragma unroll
    for (int y = 0; y < YD; ++y) u[y] = __shfl_xor(v[y], 16, 32);

    // ---- WMMA: prefix moments for ALL f, two 16-pixel B tiles ----
    // B is 32x16 f16: lanes 0-15 hold column N=nlo, K=0..15 (elem e -> K=e);
    // lanes 16-31 hold K=16..31 (elem e -> K=16+e).
#pragma unroll
    for (int pt = 0; pt < 2; ++pt) {
      v16h bh, bl, ch, cl;  // hi/lo split of v and v*v
#pragma unroll
      for (int e = 0; e < 16; ++e) {
        float x;
        if (pt == 0) x = half ? u[16 + e] : v[e];   // pixels 0..15
        else         x = half ? v[16 + e] : u[e];   // pixels 16..31
        const _Float16 xh = (_Float16)x;
        bh[e] = xh;
        bl[e] = (_Float16)(x - (float)xh);
        const float xx = x * x;
        const _Float16 xxh = (_Float16)xx;
        ch[e] = xxh;
        cl[e] = (_Float16)(xx - (float)xxh);
      }
#pragma unroll
      for (int ft = 0; ft < 2; ++ft) {
        v8f d0 = {}, d1 = {}, d2 = {};
        d0 = __builtin_amdgcn_wmma_f32_16x16x32_f16(false, a1[ft], false, bh, (short)0, d0, false, false);
        d0 = __builtin_amdgcn_wmma_f32_16x16x32_f16(false, a1[ft], false, bl, (short)0, d0, false, false);
        d1 = __builtin_amdgcn_wmma_f32_16x16x32_f16(false, a2[ft], false, bh, (short)0, d1, false, false);
        d1 = __builtin_amdgcn_wmma_f32_16x16x32_f16(false, a2[ft], false, bl, (short)0, d1, false, false);
        d2 = __builtin_amdgcn_wmma_f32_16x16x32_f16(false, a1[ft], false, ch, (short)0, d2, false, false);
        d2 = __builtin_amdgcn_wmma_f32_16x16x32_f16(false, a1[ft], false, cl, (short)0, d2, false, false);
        // D layout: VGPR r holds (M=r, N=nlo) for lanes 0-15, (M=r+8, N=nlo) for 16-31
#pragma unroll
        for (int r = 0; r < 8; ++r) {
          const int frow = 16 * ft + r + 8 * half;
          const int px   = 16 * pt + nlo;
          sP[wib][0][frow][px] = d0[r];
          sP[wib][1][frow][px] = d1[r];
          sP[wib][2][frow][px] = d2[r];
        }
      }
    }
    __syncthreads();
    const float p0 = sP[wib][0][fidx][lane];  // sum_{t<f} v
    const float p1 = sP[wib][1][fidx][lane];  // sum_{t<f} t*v
    const float p2 = sP[wib][2][fidx][lane];  // sum_{t<f} v^2
    __syncthreads();

    // ---- closed-form regression + residual per segment ----
    auto seg = [](float n, float sx, float sxx, float sy, float sxy, float syy) -> float {
      const float ns  = fmaxf(n, 1.0f);
      const float cov = sxy - sx * sy / ns;
      const float var = sxx - sx * sx / ns;
      const float vs  = (var > 0.0f) ? var : 1.0f;
      const float sl  = fminf(fmaxf(cov / vs, 0.0f), 2.0f);   // clip to [0,2]
      const float ic  = (sy - sl * sx) / ns;
      const float res = syy - 2.0f * sl * sxy - 2.0f * ic * sy
                      + sl * sl * sxx + 2.0f * sl * ic * sx + n * ic * ic;
      return (n >= 3.0f) ? res : 0.0f;                        // 'valid' gate
    };

    const float ff = (float)fidx;
    // before: x = t, t in [0,f)
    const float n_b   = ff;
    const float sx_b  = 0.5f * ff * (ff - 1.0f);
    const float sxx_b = (ff - 1.0f) * ff * (2.0f * ff - 1.0f) * (1.0f / 6.0f);
    // after: x = t - f, t in [f,32); g = 31 - f
    const float g     = 31.0f - ff;
    const float n_a   = 32.0f - ff;
    const float sx_a  = 0.5f * g * (g + 1.0f);
    const float sxx_a = g * (g + 1.0f) * (2.0f * g + 1.0f) * (1.0f / 6.0f);
    const float sy_a  = S0 - p0;
    const float sxy_a = (S1 - p1) - ff * sy_a;
    const float syy_a = S2 - p2;

    acc += (seg(n_b, sx_b, sxx_b, p0, p1, p2) +
            seg(n_a, sx_a, sxx_a, sy_a, sxy_a, syy_a)) * (1.0f / 32.0f);
  }

  // ---- wave reduction + global accumulate of the mean ----
#pragma unroll
  for (int off = 16; off >= 1; off >>= 1) acc += __shfl_xor(acc, off, 32);
  if (lane == 0) atomicAdd(lossp, acc * (1.0f / (float)NPIX));
}

extern "C" void kernel_launch(void* const* d_in, const int* in_sizes, int n_in,
                              void* d_out, int out_size, void* d_ws, size_t ws_size,
                              hipStream_t stream) {
  const float* outp = (const float*)d_in[0];   // 'out' (B,Y,H,W) f32; 'target' unused
  float* lossp = (float*)d_out;                // 1 float

  drl_zero_kernel<<<1, 32, 0, stream>>>(lossp);
  drl_main_kernel<<<GRID, BLOCK, 0, stream>>>(outp, lossp);
}
